// GCNEncoder_36180804502037
// MI455X (gfx1250) — matrix operand order repaired and checked by reading, not compiled, not run on hardware
//
#include <hip/hip_runtime.h>
#include <hip/hip_bf16.h>
#include <math.h>

typedef __attribute__((ext_vector_type(16))) _Float16 v16h;
typedef __attribute__((ext_vector_type(8)))  float    v8f;

union F16x16 { v16h v; _Float16 h[16]; unsigned int u[8]; };
union F32x8  { v8f  v; float    f[8];  };
union H4     { _Float16 h[4]; uint2 v2; };

__device__ __forceinline__ float gelu_exact(float x) {
  return 0.5f * x * (1.0f + erff(x * 0.70710678118654752f));
}

// =====================================================================
// Kernel 1: conv1 (Cin=1, K=9, s=2, p=4) + GroupNorm(8 ch/group) + GELU
// grid = 1024*8 (batch,group); block = 256. Two-pass (stats, normalize);
// LDS = 16KB x-row. h1 written as f16 with 8B-vectorized stores.
// =====================================================================
__global__ __launch_bounds__(256) void conv1_gn_gelu(
    const float* __restrict__ x, const float* __restrict__ w1,
    const float* __restrict__ g1, const float* __restrict__ b1,
    _Float16* __restrict__ h1) {
  __shared__ float xs[4096];
  __shared__ float ws1[8][9];
  __shared__ float gs, gs2, smean, srsig;
  const int blk = blockIdx.x;
  const int b = blk >> 3, g = blk & 7;
  const int tid = threadIdx.x;
  if (tid == 0) { gs = 0.f; gs2 = 0.f; }
  for (int i = tid; i < 4096; i += 256) xs[i] = x[(size_t)b * 4096 + i];
  if (tid < 72) ws1[tid / 9][tid % 9] = w1[(size_t)(8 * g) * 9 + tid];
  __syncthreads();

  float s = 0.f, s2 = 0.f;
  for (int u = 0; u < 16; ++u) {                 // 8 ch * 512 quads / 256 thr
    int id = u * 256 + tid;
    int co = id >> 9;
    int t0 = (id & 511) * 4;
    #pragma unroll
    for (int j = 0; j < 4; ++j) {
      int t = t0 + j;
      float a = 0.f;
      #pragma unroll
      for (int k = 0; k < 9; ++k) {
        int q = 2 * t + k - 4;
        float xv = (q >= 0 && q < 4096) ? xs[q] : 0.f;
        a += xv * ws1[co][k];
      }
      s += a; s2 += a * a;
    }
  }
  atomicAdd(&gs, s); atomicAdd(&gs2, s2);
  __syncthreads();
  if (tid == 0) {
    float m = gs * (1.f / 16384.f);
    float v = gs2 * (1.f / 16384.f) - m * m;
    smean = m; srsig = rsqrtf(v + 1e-5f);
  }
  __syncthreads();
  const float m = smean, rs = srsig;
  for (int u = 0; u < 16; ++u) {                 // recompute + normalize
    int id = u * 256 + tid;
    int co = id >> 9;
    int t0 = (id & 511) * 4;
    int ch = 8 * g + co;
    const float gam = g1[ch], bet = b1[ch];
    H4 pk;
    #pragma unroll
    for (int j = 0; j < 4; ++j) {
      int t = t0 + j;
      float a = 0.f;
      #pragma unroll
      for (int k = 0; k < 9; ++k) {
        int q = 2 * t + k - 4;
        float xv = (q >= 0 && q < 4096) ? xs[q] : 0.f;
        a += xv * ws1[co][k];
      }
      float y = (a - m) * rs * gam + bet;
      pk.h[j] = (_Float16)gelu_exact(y);
    }
    *(uint2*)(h1 + ((size_t)b * 64 + ch) * 2048 + t0) = pk.v2;  // 8B aligned
  }
}

// =====================================================================
// Kernel 2: conv2 (64->64, K=7, s=2, p=3) as implicit GEMM via
// v_wmma_f32_16x16x32_f16 with K padded to 8 taps (left zero tap, offsets
// -4..+3): K_tot = 512 = 16 chunks; every B-fragment half is one
// contiguous, 4B-aligned 8-half slab run -> ds_load_2addr_b32 x4 per WMMA.
// A fragments are pre-swizzled once per block into a 16KB LDS table
// (32B/lane, aligned -> 2x ds_load_b128 per WMMA) instead of pinning
// 128 VGPRs, freeing registers for load/WMMA overlap and occupancy.
// Fused GroupNorm + GELU + temporal mean pool, two-pass (stats, apply).
// grid = 1024*4 (batch, 16-Cout tile); block = 256 (8 waves, 16 cols each).
// =====================================================================
__device__ __forceinline__ void load_slab(_Float16 (*slab)[272],
                                          const _Float16* __restrict__ h1b,
                                          int tt, int tid) {
  const int base_tp = tt * 256 - 8;              // even base: t' = base + col
  for (int i = tid; i < 64 * 34; i += 256) {     // 34 chunks of 8 halves
    int ci = i / 34, c = i - ci * 34;
    int tp0 = base_tp + 8 * c;
    if (tp0 >= 0 && tp0 <= 2040) {
      uint4 v = *(const uint4*)(h1b + (size_t)ci * 2048 + tp0);  // 16B aligned
      *(uint4*)&slab[ci][8 * c] = v;                             // 16B aligned
    } else {
      #pragma unroll
      for (int j = 0; j < 8; ++j) {
        int tp = tp0 + j;
        _Float16 v = (_Float16)0.f;
        if (tp >= 0 && tp < 2048) v = h1b[(size_t)ci * 2048 + tp];
        slab[ci][8 * c + j] = v;
      }
    }
  }
}

__device__ __forceinline__ void load_bfrag(F16x16& bf,
                                           const _Float16 (*slab)[272],
                                           int kb, int col0, int half) {
  // B layout used here: lanes 0-15 hold k=0..15, lanes 16-31 hold k=16..31.
  // logical kp = kb*32 + e + 16*half; ci = kp>>3 -> runs ci0, ci0+1.
  const int ci0 = 4 * kb + 2 * half;
  const unsigned int* p0 = (const unsigned int*)&slab[ci0][col0];
  const unsigned int* p1 = (const unsigned int*)&slab[ci0 + 1][col0];
  #pragma unroll
  for (int j = 0; j < 4; ++j) { bf.u[j] = p0[j]; bf.u[4 + j] = p1[j]; }
}

__device__ __forceinline__ v8f conv2_tile(const _Float16 (*wfrag)[32][16],
                                          const _Float16 (*slab)[272],
                                          int wv, int nloc, int half, int lane) {
  F32x8 z;
  #pragma unroll
  for (int r = 0; r < 8; ++r) z.f[r] = 0.f;
  v8f acc = z.v;
  // column t = t0 + 16*wv + nloc; tap run starts at t' = 2t-4 ->
  // col0 = (2t-4) - (2*t0-8) = 32*wv + 2*nloc + 4 (even -> 4B aligned)
  const int col0 = 32 * wv + 2 * nloc + 4;
  // distance-1 pipeline over the 16 K-chunks (A from LDS table, B from slab)
  F16x16 bf[2];
  v16h   af[2];
  load_bfrag(bf[0], slab, 0, col0, half);
  af[0] = *(const v16h*)&wfrag[0][lane][0];      // 32B aligned -> b128 x2
  #pragma unroll
  for (int kb = 0; kb < 16; ++kb) {
    if (kb + 1 < 16) {
      load_bfrag(bf[(kb + 1) & 1], slab, kb + 1, col0, half);
      af[(kb + 1) & 1] = *(const v16h*)&wfrag[kb + 1][lane][0];
    }
    acc = __builtin_amdgcn_wmma_f32_16x16x32_f16(false, af[kb & 1], false,
                                                 bf[kb & 1].v, (short)0, acc,
                                                 false, false);
  }
  return acc;
}

__global__ __launch_bounds__(256) void conv2_gn_gelu_pool(
    const _Float16* __restrict__ h1, const float* __restrict__ w2,
    const float* __restrict__ g2, const float* __restrict__ b2,
    float* __restrict__ hmean) {
  __shared__ _Float16 slab[64][272];             // 34816 B
  __shared__ _Float16 wfrag[16][32][16];         // 16384 B, 32B per lane slot
  __shared__ float gsum[2], gsumsq[2], gmean[2], grsig[2];
  __shared__ float chsum[16];
  const int blk = blockIdx.x;
  const int b = blk >> 2, c2 = blk & 3;
  const int tid = threadIdx.x;
  const int lane = tid & 31, wv = tid >> 5;
  const int half = lane >> 4, nloc = lane & 15;
  if (tid < 2) { gsum[tid] = 0.f; gsumsq[tid] = 0.f; }
  if (tid < 16) chsum[tid] = 0.f;

  // Build A-fragment table cooperatively: fragment (kb, lane) holds this code's
  // A layout (lanes0-15 K 0-15, lanes16-31 K 16-31) for logical
  // kp = ci*8 + ko, ko=0 the zero pad tap (offset -4), ko=1..7 -> taps 0..6.
  for (int i = tid; i < 16 * 32; i += 256) {     // i = kb*32 + dest lane
    int kb = i >> 5, ln = i & 31;
    int hf = ln >> 4, row = ln & 15;
    const float* wrow = w2 + (size_t)(16 * c2 + row) * 448;
    #pragma unroll
    for (int e = 0; e < 16; ++e) {
      int kp = kb * 32 + e + 16 * hf;
      int ci = kp >> 3, ko = kp & 7;
      wfrag[kb][ln][e] = (ko == 0) ? (_Float16)0.f
                                   : (_Float16)wrow[ci * 7 + ko - 1];
    }
  }
  const _Float16* h1b = h1 + (size_t)b * 64 * 2048;

  // ---- pass 1: conv via WMMA, accumulate group stats
  float s = 0.f, s2 = 0.f;
  for (int tt = 0; tt < 8; ++tt) {
    __syncthreads();
    load_slab(slab, h1b, tt, tid);
    if (tt < 7) __builtin_prefetch(h1b + (size_t)(tid & 63) * 2048 + tt * 256 + 248, 0, 1);
    __syncthreads();
    F32x8 ua; ua.v = conv2_tile(wfrag, slab, wv, nloc, half, lane);
    #pragma unroll
    for (int r = 0; r < 8; ++r) { s += ua.f[r]; s2 += ua.f[r] * ua.f[r]; }
  }
  // D-tile rows: half=0 -> channels 0..7 (group 2*c2), half=1 -> 8..15
  atomicAdd(&gsum[half], s); atomicAdd(&gsumsq[half], s2);
  __syncthreads();
  if (tid < 2) {
    float m = gsum[tid] * (1.f / 8192.f);
    float v = gsumsq[tid] * (1.f / 8192.f) - m * m;
    gmean[tid] = m; grsig[tid] = rsqrtf(v + 1e-5f);
  }
  __syncthreads();
  const float m = gmean[half], rs = grsig[half];
  float gam[8], bet[8], racc[8];
  #pragma unroll
  for (int r = 0; r < 8; ++r) {
    int ch = 16 * c2 + r + 8 * half;             // channel of VGPR row r
    gam[r] = g2[ch]; bet[r] = b2[ch]; racc[r] = 0.f;
  }
  // ---- pass 2: recompute, normalize, GELU, pool over t
  for (int tt = 0; tt < 8; ++tt) {
    __syncthreads();
    load_slab(slab, h1b, tt, tid);
    __syncthreads();
    F32x8 ua; ua.v = conv2_tile(wfrag, slab, wv, nloc, half, lane);
    #pragma unroll
    for (int r = 0; r < 8; ++r) {
      float y = (ua.f[r] - m) * rs * gam[r] + bet[r];
      racc[r] += gelu_exact(y);
    }
  }
  #pragma unroll
  for (int r = 0; r < 8; ++r) atomicAdd(&chsum[r + 8 * half], racc[r]);
  __syncthreads();
  if (tid < 16)
    hmean[(size_t)b * 64 + 16 * c2 + tid] = chsum[tid] * (1.f / 1024.f);
}

// =====================================================================
// Kernel 3: projection + normalized-adjacency GCN (N=4, D=128) + LNs.
// grid = 256 (batch); block = 128 (one thread per feature d).
// =====================================================================
__global__ __launch_bounds__(128) void gcn_head(
    const float* __restrict__ hmean, const float* __restrict__ pw,
    const float* __restrict__ pb, const float* __restrict__ A,
    const float* __restrict__ l1w, const float* __restrict__ l1g,
    const float* __restrict__ l1b, const float* __restrict__ l2w,
    const float* __restrict__ l2g, const float* __restrict__ l2b,
    const float* __restrict__ og, const float* __restrict__ ob,
    float* __restrict__ out) {
  __shared__ float hm[4][64];
  __shared__ float Hs[4][128];
  __shared__ float red[128];
  __shared__ float Ahat[16];
  const int b = blockIdx.x, d = threadIdx.x;
  for (int i = d; i < 256; i += 128) hm[i >> 6][i & 63] = hmean[(size_t)b * 256 + i];
  if (d == 0) {
    float At[16], dis[4];
    for (int n = 0; n < 4; ++n)
      for (int mm = 0; mm < 4; ++mm) At[n * 4 + mm] = A[n * 4 + mm] + (n == mm ? 1.f : 0.f);
    for (int n = 0; n < 4; ++n) {
      float sdeg = 0.f;
      for (int mm = 0; mm < 4; ++mm) sdeg += At[n * 4 + mm];
      dis[n] = rsqrtf(sdeg + 1e-6f);
    }
    for (int n = 0; n < 4; ++n)
      for (int mm = 0; mm < 4; ++mm) Ahat[n * 4 + mm] = dis[n] * At[n * 4 + mm] * dis[mm];
  }
  __syncthreads();
  float H[4];
  #pragma unroll
  for (int c = 0; c < 4; ++c) {
    float a = pb[d];
    for (int j = 0; j < 64; ++j) a += hm[c][j] * pw[(size_t)d * 64 + j];
    H[c] = a;
  }
  const float* lw_arr[2] = { l1w, l2w };
  const float* lg_arr[2] = { l1g, l2g };
  const float* lb_arr[2] = { l1b, l2b };
  #pragma unroll
  for (int L = 0; L < 2; ++L) {
    float Hmix[4];
    #pragma unroll
    for (int n = 0; n < 4; ++n) {
      float a = 0.f;
      #pragma unroll
      for (int mm = 0; mm < 4; ++mm) a += Ahat[n * 4 + mm] * H[mm];
      Hmix[n] = a;
    }
    __syncthreads();
    #pragma unroll
    for (int n = 0; n < 4; ++n) Hs[n][d] = Hmix[n];
    __syncthreads();
    const float* lw = lw_arr[L];
    #pragma unroll
    for (int n = 0; n < 4; ++n) {
      float y = 0.f;
      for (int j = 0; j < 128; ++j) y += Hs[n][j] * lw[(size_t)d * 128 + j];
      red[d] = y; __syncthreads();
      for (int st = 64; st > 0; st >>= 1) { if (d < st) red[d] += red[d + st]; __syncthreads(); }
      float mean = red[0] * (1.f / 128.f); __syncthreads();
      float c0 = y - mean;
      red[d] = c0 * c0; __syncthreads();
      for (int st = 64; st > 0; st >>= 1) { if (d < st) red[d] += red[d + st]; __syncthreads(); }
      float var = red[0] * (1.f / 128.f); __syncthreads();
      float z = c0 * rsqrtf(var + 1e-5f) * lg_arr[L][d] + lb_arr[L][d];
      H[n] = gelu_exact(z);
    }
  }
  float o = 0.f;
  #pragma unroll
  for (int n = 0; n < 4; ++n) {
    float y = H[n];
    red[d] = y; __syncthreads();
    for (int st = 64; st > 0; st >>= 1) { if (d < st) red[d] += red[d + st]; __syncthreads(); }
    float mean = red[0] * (1.f / 128.f); __syncthreads();
    float c0 = y - mean;
    red[d] = c0 * c0; __syncthreads();
    for (int st = 64; st > 0; st >>= 1) { if (d < st) red[d] += red[d + st]; __syncthreads(); }
    float var = red[0] * (1.f / 128.f); __syncthreads();
    o += c0 * rsqrtf(var + 1e-5f) * og[d] + ob[d];
  }
  out[(size_t)b * 128 + d] = o * 0.25f;
}

// =====================================================================
extern "C" void kernel_launch(void* const* d_in, const int* in_sizes, int n_in,
                              void* d_out, int out_size, void* d_ws, size_t ws_size,
                              hipStream_t stream) {
  (void)in_sizes; (void)n_in; (void)out_size; (void)ws_size;
  const float* x   = (const float*)d_in[0];
  const float* c1w = (const float*)d_in[1];
  const float* g1g = (const float*)d_in[2];
  const float* g1b = (const float*)d_in[3];
  const float* c2w = (const float*)d_in[4];
  const float* g2g = (const float*)d_in[5];
  const float* g2b = (const float*)d_in[6];
  const float* pw  = (const float*)d_in[7];
  const float* pb  = (const float*)d_in[8];
  const float* A   = (const float*)d_in[9];
  const float* l1w = (const float*)d_in[10];
  const float* l1g = (const float*)d_in[11];
  const float* l1b = (const float*)d_in[12];
  const float* l2w = (const float*)d_in[13];
  const float* l2g = (const float*)d_in[14];
  const float* l2b = (const float*)d_in[15];
  const float* og  = (const float*)d_in[16];
  const float* ob  = (const float*)d_in[17];
  float* out = (float*)d_out;

  // workspace layout: h1 (1024 x 64 x 2048 f16 = 256 MiB), then hmean (1024x64 f32)
  _Float16* h1 = (_Float16*)d_ws;
  float* hmean = (float*)((char*)d_ws + (size_t)1024 * 64 * 2048 * sizeof(_Float16));

  conv1_gn_gelu<<<8192, 256, 0, stream>>>(x, c1w, g1g, g1b, h1);
  conv2_gn_gelu_pool<<<4096, 256, 0, stream>>>(h1, c2w, g2g, g2b, hmean);
  gcn_head<<<256, 128, 0, stream>>>(hmean, pw, pb, A, l1w, l1g, l1b,
                                    l2w, l2g, l2b, og, ob, out);
}
